// DualModel3_19559281066670
// MI455X (gfx1250) — matrix operand order repaired and checked by reading, not compiled, not run on hardware
//
#include <hip/hip_runtime.h>

// ---------------------------------------------------------------------------
// MI455X (gfx1250).
// x1 = x @ fcW^T + fcb                                 [256,1000]
// x2 = (leakyrelu(x @ W1 + b1) * W2).sum(h) + b2       [256,4096]
//
// Roofline: W1 (134MB fp32) dominates -> ~6.5us HBM floor @23.3TB/s; bf16
// WMMA keeps compute above that roof. x/fcW pre-converted to bf16 in d_ws
// (zero conversion VALU in GEMM loops); W1 converted exactly once while
// staged through LDS. mlp_gemm N-blocked 16 decoders/block so the bf16 x
// copy is re-read only 256x from L2 (~40TB/s) instead of 1024x.
// ---------------------------------------------------------------------------

typedef __attribute__((ext_vector_type(16))) __bf16          v16bf;
typedef __attribute__((ext_vector_type(8)))  __bf16          v8bf;
typedef __attribute__((ext_vector_type(8)))  unsigned short  v8u;
typedef __attribute__((ext_vector_type(8)))  float           v8f;

#define B_   256
#define IN_  2048
#define NC_  1000
#define O_   4096
#define H_   4
#define NEG_SLOPE 0.1f

// fp32 -> bf16 round-to-nearest-even
__device__ __forceinline__ unsigned short f2bf(float f) {
  unsigned u = __builtin_bit_cast(unsigned, f);
  unsigned r = u + 0x7FFFu + ((u >> 16) & 1u);
  return (unsigned short)(r >> 16);
}

__device__ __forceinline__ v8u pack8(const float4 a, const float4 b) {
  v8u u;
  u[0] = f2bf(a.x); u[1] = f2bf(a.y); u[2] = f2bf(a.z); u[3] = f2bf(a.w);
  u[4] = f2bf(b.x); u[5] = f2bf(b.y); u[6] = f2bf(b.z); u[7] = f2bf(b.w);
  return u;
}

__device__ __forceinline__ v8f wmma_bf16(v16bf a, v16bf b, v8f c) {
  return __builtin_amdgcn_wmma_f32_16x16x32_bf16(false, a, false, b,
                                                 (short)0, c, false, false);
}

// A-fragment (16x32): per-lane k = 16*hi+0..15 (contiguous), raw bf16 loads.
__device__ __forceinline__ v16bf load_afrag_bf(const __bf16* __restrict__ rp,
                                               int ko, int hi) {
  v8bf lo = *(const v8bf*)(rp + ko + 16 * hi);
  v8bf hh = *(const v8bf*)(rp + ko + 16 * hi + 8);
  return __builtin_shufflevector(lo, hh, 0, 1, 2, 3, 4, 5, 6, 7,
                                 8, 9, 10, 11, 12, 13, 14, 15);
}

// ---------------------------------------------------------------------------
// Prep: fp32 -> bf16, 8 elements/thread.
// ---------------------------------------------------------------------------
__global__ __launch_bounds__(256) void cvt_fp32_bf16(
    const float* __restrict__ src, v8u* __restrict__ dst, int n8) {
  const int i = blockIdx.x * 256 + threadIdx.x;
  if (i >= n8) return;
  const float4* s = (const float4*)src + (size_t)i * 2;
  dst[i] = pack8(s[0], s[1]);
}

// ---------------------------------------------------------------------------
// Kernel 1: classifier head. One wave per 16x16 tile (16 x 63 tiles).
// ---------------------------------------------------------------------------
__global__ __launch_bounds__(256) void head_gemm(
    const __bf16* __restrict__ xb, const __bf16* __restrict__ fcWb,
    const float* __restrict__ fcb, float* __restrict__ out) {
  const int lane = threadIdx.x & 31;
  const int wid  = blockIdx.x * 8 + (threadIdx.x >> 5);
  const int mt = wid & 15;
  const int nt = wid >> 4;                     // 63 N tiles (1000 = 62*16+8)
  const int m0 = mt * 16;
  const int n0 = nt * 16;
  const int c  = lane & 15;
  const int hi = lane >> 4;

  const int brow = (n0 + c < NC_) ? (n0 + c) : (NC_ - 1);   // clamp loads
  const __bf16* ap = xb   + (size_t)(m0 + c) * IN_;
  const __bf16* bp = fcWb + (size_t)brow * IN_;

  v8f acc = {};
  for (int kb = 0; kb < IN_; kb += 32) {
    v16bf afrag = load_afrag_bf(ap, kb, hi);
    v16bf bfrag = *(const v16bf*)(bp + kb + 16 * hi);   // 16 contiguous k
    acc = wmma_bf16(afrag, bfrag, acc);
  }

  const int n = n0 + c;
  if (n < NC_) {
    const float bias = fcb[n];
#pragma unroll
    for (int r = 0; r < 8; ++r) {              // C/D: vgpr r -> m = r + 8*hi
      const int m = m0 + r + 8 * hi;
      out[(size_t)m * NC_ + n] = acc[r] + bias;
    }
  }
}

// ---------------------------------------------------------------------------
// Kernel 2: grouped MLP GEMM + fused epilogue.
// 256 blocks x 8 waves; block owns 16 decoders (64 columns), sweeps all
// M=256 -> W1 streamed from HBM exactly once; x re-read 256x from L2.
// Wave tile: 2 M-tiles x 4 N-tiles (8 f32 accumulators, 16 WMMA / 64-k chunk).
// Staging: thread owns two (o, k-pair) tasks -> 2x contiguous b128 loads,
// bf16 pair packed per column -> 4x ds_store_b32 per task. Double-buffered
// LDS, one barrier per chunk; global_prefetch_b8 issued 8 chunks ahead.
// ---------------------------------------------------------------------------
__global__ __launch_bounds__(256) void mlp_gemm(
    const __bf16* __restrict__ xb, const float* __restrict__ W1,
    const float* __restrict__ b1,  const float* __restrict__ W2,
    const float* __restrict__ b2,  float* __restrict__ x2) {
  // [buf][col][kpair] : 64 cols x 32 dwords (= 64 k of bf16, column-major)
  __shared__ __align__(32) unsigned int ldsB32[2][64 * 32];   // 16KB total

  const int tid   = threadIdx.x;
  const int lane  = tid & 31;
  const int w     = tid >> 5;             // wave 0..7
  const int obase = blockIdx.x * 16;      // 16 decoders per block
  const int c     = lane & 15;
  const int hi    = lane >> 4;

  const __bf16* apA = xb + (size_t)(32 * w + c)      * IN_;
  const __bf16* apB = xb + (size_t)(32 * w + 16 + c) * IN_;

  // Staging tasks: task tau in [0,512): orel = tau>>5, kpair = tau&31.
  // Thread handles tau = tid and tid+256 (same kpair, orel+8).
  const int orel = tid >> 5;              // 0..7
  const int kp   = tid & 31;              // k-pair index, k0 = 2*kp
  const float* wp0 = W1 + (size_t)(obase + orel)     * (IN_ * H_) + (kp * 2) * H_;
  const float* wp1 = W1 + (size_t)(obase + orel + 8) * (IN_ * H_) + (kp * 2) * H_;
  const int cb0 = orel * 4;               // column base of task 0
  const int cb1 = (orel + 8) * 4;         // column base of task 1

  v8f acc[4][2] = {};
  for (int kb = 0, it = 0; kb < IN_; kb += 64, ++it) {
    const int buf = it & 1;
    // ---- stage W1 slab: 16 o x 64 k, fp32 -> packed bf16 pairs ----
    const float4* p0 = (const float4*)(wp0 + (size_t)kb * H_);
    const float4* p1 = (const float4*)(wp1 + (size_t)kb * H_);
    float4 a0 = p0[0], b0 = p0[1];        // k0, k0+1 (all four h)
    float4 a1 = p1[0], b1v = p1[1];
    ldsB32[buf][(cb0 + 0) * 32 + kp] = (unsigned)f2bf(a0.x) | ((unsigned)f2bf(b0.x) << 16);
    ldsB32[buf][(cb0 + 1) * 32 + kp] = (unsigned)f2bf(a0.y) | ((unsigned)f2bf(b0.y) << 16);
    ldsB32[buf][(cb0 + 2) * 32 + kp] = (unsigned)f2bf(a0.z) | ((unsigned)f2bf(b0.z) << 16);
    ldsB32[buf][(cb0 + 3) * 32 + kp] = (unsigned)f2bf(a0.w) | ((unsigned)f2bf(b0.w) << 16);
    ldsB32[buf][(cb1 + 0) * 32 + kp] = (unsigned)f2bf(a1.x) | ((unsigned)f2bf(b1v.x) << 16);
    ldsB32[buf][(cb1 + 1) * 32 + kp] = (unsigned)f2bf(a1.y) | ((unsigned)f2bf(b1v.y) << 16);
    ldsB32[buf][(cb1 + 2) * 32 + kp] = (unsigned)f2bf(a1.z) | ((unsigned)f2bf(b1v.z) << 16);
    ldsB32[buf][(cb1 + 3) * 32 + kp] = (unsigned)f2bf(a1.w) | ((unsigned)f2bf(b1v.w) << 16);
    if (kb + 512 < IN_) {                 // stream W1 ahead (8 chunks)
      __builtin_prefetch(wp0 + (size_t)(kb + 512) * H_, 0, 1);
      __builtin_prefetch(wp1 + (size_t)(kb + 512) * H_, 0, 1);
    }
    __syncthreads();                      // dbl-buffer -> 1 barrier/iter

    const unsigned short* lb = (const unsigned short*)ldsB32[buf];
#pragma unroll
    for (int s = 0; s < 2; ++s) {         // two K=32 WMMA steps
      const int ko = kb + s * 32;
      v16bf af0 = load_afrag_bf(apA, ko, hi);
      v16bf af1 = load_afrag_bf(apB, ko, hi);
#pragma unroll
      for (int n = 0; n < 4; ++n) {       // 4 N-tiles, B fragment reused 2x
        const v16bf bfrag =
            *(const v16bf*)&lb[(n * 16 + c) * 64 + s * 32 + hi * 16];
        acc[n][0] = wmma_bf16(af0, bfrag, acc[n][0]);
        acc[n][1] = wmma_bf16(af1, bfrag, acc[n][1]);
      }
    }
  }

  // Epilogue: +b1, leaky-relu, *W2, quad-shuffle reduce over h, +b2.
#pragma unroll
  for (int n = 0; n < 4; ++n) {
    const int colg = obase * 4 + n * 16;           // global o*4+h base
    const int o    = obase + n * 4 + (c >> 2);
    const float w2v = W2[colg + c];
    const float bv1 = b1[colg + c];
    const float bv2 = b2[o];
#pragma unroll
    for (int t = 0; t < 2; ++t) {
#pragma unroll
      for (int r = 0; r < 8; ++r) {
        float v = acc[n][t][r] + bv1;
        v = (v >= 0.f) ? v : NEG_SLOPE * v;
        float s = v * w2v;
        s += __shfl_xor(s, 1, 32);
        s += __shfl_xor(s, 2, 32);
        if ((c & 3) == 0) {
          const int m = 32 * w + t * 16 + r + 8 * hi;
          x2[(size_t)m * O_ + o] = s + bv2;
        }
      }
    }
  }
}

extern "C" void kernel_launch(void* const* d_in, const int* in_sizes, int n_in,
                              void* d_out, int out_size, void* d_ws, size_t ws_size,
                              hipStream_t stream) {
  const float* x   = (const float*)d_in[0];
  const float* fcW = (const float*)d_in[1];
  const float* fcb = (const float*)d_in[2];
  const float* W1  = (const float*)d_in[3];
  const float* b1  = (const float*)d_in[4];
  const float* W2  = (const float*)d_in[5];
  const float* b2  = (const float*)d_in[6];
  float* out = (float*)d_out;

  // Workspace: [0,1MB) x as bf16 ; [1MB,5MB) fcW as bf16
  __bf16* xb   = (__bf16*)d_ws;
  __bf16* fcWb = (__bf16*)((char*)d_ws + (size_t)B_ * IN_ * sizeof(__bf16));

  cvt_fp32_bf16<<<(B_ * IN_ / 8 + 255) / 256, 256, 0, stream>>>(x, (v8u*)xb, B_ * IN_ / 8);
  cvt_fp32_bf16<<<(NC_ * IN_ / 8 + 255) / 256, 256, 0, stream>>>(fcW, (v8u*)fcWb, NC_ * IN_ / 8);

  head_gemm<<<126, 256, 0, stream>>>(xb, fcWb, fcb, out);
  mlp_gemm<<<256, 256, 0, stream>>>(xb, W1, b1, W2, b2, out + (size_t)B_ * NC_);
}
